// Pruner_4372276707790
// MI455X (gfx1250) — matrix operand order repaired and checked by reading, not compiled, not run on hardware
//
#include <hip/hip_runtime.h>
#include <hip/hip_bf16.h>

typedef __attribute__((ext_vector_type(16))) __bf16 v16bf;
typedef __attribute__((ext_vector_type(8)))  float  v8f;

#define B_   16
#define N_   4096
#define D_   1024
#define H_   1024
#define K_   1024          // beam size
#define NEGV (-1e20f)

__device__ __forceinline__ unsigned short f32_to_bf16(float f) {
    unsigned int u = __float_as_uint(f);
    unsigned int r = (u + 0x7FFFu + ((u >> 16) & 1u)) >> 16;   // RNE
    return (unsigned short)r;
}

// ---------------------------------------------------------------------------
// Kernel 0: W1 (D x H, f32 row-major) -> bf16, tiled into B-fragment order.
// Tile = (ktile 32) x (htile 16). tileIdx = ktile*64 + htile.
// Within tile: lane's 16 elements contiguous; lane l: N = l&15,
// K = 8*(l>=16) + (i&7) + 16*(i>=8)  (i = element 0..15)  [ISA 7.12.2 16-bit B layout]
// ---------------------------------------------------------------------------
__global__ void convert_w1_kernel(const float* __restrict__ W1,
                                  unsigned short* __restrict__ w1bf) {
    int e = blockIdx.x * blockDim.x + threadIdx.x;
    if (e >= D_ * H_) return;
    int tile   = e >> 9;          // /512
    int within = e & 511;
    int lane   = within >> 4;
    int i      = within & 15;
    int ktile  = tile >> 6;       // /64
    int htile  = tile & 63;
    int n      = lane & 15;
    int koff   = ((lane >> 4) << 3) + (i & 7) + ((i >> 3) << 4);
    int k      = ktile * 32 + koff;
    int h      = htile * 16 + n;
    w1bf[e] = f32_to_bf16(W1[k * H_ + h]);
}

// ---------------------------------------------------------------------------
// Kernel 1: fused scorer. One block = 32 rows (two 16-row A tiles) of the
// flattened (B*N, D) E matrix. Each B fragment feeds TWO wmma's (acc0/acc1):
// halves L2 traffic for W1 and gives two independent accumulation chains.
// scores[row] = mask ? (relu(E_row . W1 + b1) . w2 + b2) : -1e20
// ---------------------------------------------------------------------------
__global__ void __launch_bounds__(256)
scorer_kernel(const float* __restrict__ emb,
              const int*   __restrict__ mask,
              const float* __restrict__ b1,
              const float* __restrict__ w2,
              const float* __restrict__ b2,
              const unsigned short* __restrict__ w1bf,
              float* __restrict__ scoresOut) {
    // Two A-fragment tiles: 2 * (32 k-steps * 32 lanes * 16 bf16) = 64 KB
    __shared__ __align__(32) unsigned short ldsA[32 * D_];
    __shared__ float ldsPart[8][32];

    const int t    = threadIdx.x;
    const int wave = t >> 5;
    const int lane = t & 31;
    const int rowBase = blockIdx.x * 32;

    // ---- stage 32 E rows to LDS as bf16 in A-fragment order ---------------
    // A 16-bit 16x32 layout: lanes 0-15 row M=l: elems 0-7 -> K 0-7,
    // elems 8-15 -> K 16-23; lanes 16-31 row M=l-16: K 8-15 / 24-31.
    for (int e = t; e < 32 * D_; e += 256) {
        int r = e >> 10;              // row in block tile (0..31)
        int c = e & (D_ - 1);         // K index
        float g = emb[(size_t)(rowBase + r) * D_ + c];
        int mt   = r >> 4;            // which 16-row A tile
        int rr   = r & 15;
        int kt   = c >> 5;
        int koff = c & 31;
        int half = (koff >> 3) & 1;
        int i    = (koff & 7) | ((koff >> 4) << 3);
        int ln   = rr + (half << 4);
        ldsA[mt * (16 * D_) + kt * 512 + ln * 16 + i] = f32_to_bf16(g);
    }
    __syncthreads();

    // ---- GEMM + fused relu/b1/w2 ------------------------------------------
    float spart0[8], spart1[8];
#pragma unroll
    for (int r = 0; r < 8; ++r) { spart0[r] = 0.0f; spart1[r] = 0.0f; }

    const unsigned short* ldsA0 = &ldsA[lane * 16];
    const unsigned short* ldsA1 = &ldsA[16 * D_ + lane * 16];

    for (int j = 0; j < 8; ++j) {
        const int htile = wave * 8 + j;
        const int h0    = htile * 16;
        const unsigned short* bBase = &w1bf[(size_t)htile * 512 + lane * 16];
        v8f acc0 = {0.f, 0.f, 0.f, 0.f, 0.f, 0.f, 0.f, 0.f};
        v8f acc1 = {0.f, 0.f, 0.f, 0.f, 0.f, 0.f, 0.f, 0.f};
#pragma unroll 4
        for (int kt = 0; kt < 32; ++kt) {
            v16bf bb = *(const v16bf*)&bBase[(size_t)kt * (64 * 512)];
            v16bf a0 = *(const v16bf*)&ldsA0[kt * 512];
            v16bf a1 = *(const v16bf*)&ldsA1[kt * 512];
            acc0 = __builtin_amdgcn_wmma_f32_16x16x32_bf16(
                false, a0, false, bb, (short)0, acc0, false, false);
            acc1 = __builtin_amdgcn_wmma_f32_16x16x32_bf16(
                false, a1, false, bb, (short)0, acc1, false, false);
        }
        const int n     = lane & 15;
        const float b1v = b1[h0 + n];
        const float w2v = w2[h0 + n];
#pragma unroll
        for (int r = 0; r < 8; ++r) {
            float v0 = acc0[r] + b1v;
            float v1 = acc1[r] + b1v;
            v0 = v0 > 0.0f ? v0 : 0.0f;   // relu
            v1 = v1 > 0.0f ? v1 : 0.0f;
            spart0[r] += v0 * w2v;
            spart1[r] += v1 * w2v;
        }
    }

    // ---- reduce over N within each 16-lane half (rows r and r+8) ----------
#pragma unroll
    for (int r = 0; r < 8; ++r) {
        float v0 = spart0[r];
        v0 += __shfl_xor(v0, 8, 32);
        v0 += __shfl_xor(v0, 4, 32);
        v0 += __shfl_xor(v0, 2, 32);
        v0 += __shfl_xor(v0, 1, 32);
        float v1 = spart1[r];
        v1 += __shfl_xor(v1, 8, 32);
        v1 += __shfl_xor(v1, 4, 32);
        v1 += __shfl_xor(v1, 2, 32);
        v1 += __shfl_xor(v1, 1, 32);
        if ((lane & 15) == 0) {
            int rowSub = ((lane >> 4) << 3) + r;       // lane0->r, lane16->r+8
            ldsPart[wave][rowSub]      = v0;           // rows 0..15
            ldsPart[wave][16 + rowSub] = v1;           // rows 16..31
        }
    }
    __syncthreads();

    if (t < 32) {
        float s = 0.0f;
#pragma unroll
        for (int w = 0; w < 8; ++w) s += ldsPart[w][t];
        s += b2[0];
        int row = rowBase + t;
        scoresOut[row] = (mask[row] > 0) ? s : NEGV;
    }
}

// ---------------------------------------------------------------------------
// Kernel 2: per-batch top-k (k=1024 of N=4096) via bitonic sort in LDS.
// Sort (score desc, index asc on ties), then sort top-k indices ascending.
// ---------------------------------------------------------------------------
__global__ void __launch_bounds__(512)
topk_kernel(const float* __restrict__ scores,
            const int*   __restrict__ mask,
            float* __restrict__ outMask,
            float* __restrict__ outIdx,
            float* __restrict__ outScores,
            float* __restrict__ outNk,
            int*   __restrict__ idxWs) {
    __shared__ float skey[N_];
    __shared__ int   sidx[N_];
    const int b = blockIdx.x;
    const int t = threadIdx.x;
    const int T = blockDim.x;

    for (int i = t; i < N_; i += T) { skey[i] = scores[b * N_ + i]; sidx[i] = i; }
    __syncthreads();

    // bitonic sort descending (ties: lower index first, like lax.top_k)
    for (int size = 2; size <= N_; size <<= 1) {
        for (int stride = size >> 1; stride > 0; stride >>= 1) {
            for (int i = t; i < N_ / 2; i += T) {
                int j = ((i / stride) * (stride << 1)) + (i % stride);
                int l = j + stride;
                bool descBlock = ((j & size) == 0);
                float a = skey[j], c = skey[l];
                int  ia = sidx[j], ic = sidx[l];
                bool aFirst = (a > c) || (a == c && ia < ic);
                if (descBlock ? (!aFirst) : aFirst) {
                    skey[j] = c; skey[l] = a;
                    sidx[j] = ic; sidx[l] = ia;
                }
            }
            __syncthreads();
        }
    }

    // sort top-K_ indices ascending (jnp.sort of top_indices)
    for (int size = 2; size <= K_; size <<= 1) {
        for (int stride = size >> 1; stride > 0; stride >>= 1) {
            for (int i = t; i < K_ / 2; i += T) {
                int j = ((i / stride) * (stride << 1)) + (i % stride);
                int l = j + stride;
                bool ascBlock = ((j & size) == 0);
                int ia = sidx[j], ic = sidx[l];
                bool aGreater = ia > ic;
                if (ascBlock ? aGreater : !aGreater) {
                    sidx[j] = ic; sidx[l] = ia;
                }
            }
            __syncthreads();
        }
    }

    for (int i = t; i < K_; i += T) {
        int idx = sidx[i];
        idxWs    [b * K_ + i] = idx;
        outIdx   [b * K_ + i] = (float)idx;
        outScores[b * K_ + i] = scores[b * N_ + idx];   // masked scores, as in ref
        outMask  [b * K_ + i] = (mask[b * N_ + idx] > 0) ? 1.0f : 0.0f;
    }
    if (t == 0) outNk[b] = (float)K_;
}

// ---------------------------------------------------------------------------
// Kernel 3: gather top embeddings. One block per (b, slot); float4 copies.
// ---------------------------------------------------------------------------
__global__ void __launch_bounds__(256)
gather_kernel(const float* __restrict__ emb,
              const int*   __restrict__ idxWs,
              float* __restrict__ outEmb) {
    int row = blockIdx.x;               // 0 .. B_*K_-1
    int b   = row >> 10;                // /K_
    int idx = idxWs[row];
    const float4* src = (const float4*)(emb + (size_t)(b * N_ + idx) * D_);
    float4*       dst = (float4*)(outEmb + (size_t)row * D_);
    dst[threadIdx.x] = src[threadIdx.x];   // 256 threads * float4 = 1024 floats
}

// ---------------------------------------------------------------------------
extern "C" void kernel_launch(void* const* d_in, const int* in_sizes, int n_in,
                              void* d_out, int out_size, void* d_ws, size_t ws_size,
                              hipStream_t stream) {
    const float* emb  = (const float*)d_in[0];
    const int*   mask = (const int*)  d_in[1];
    // d_in[2] = num_items_to_keep (static 1024)
    const float* W1   = (const float*)d_in[3];
    const float* b1   = (const float*)d_in[4];
    const float* w2   = (const float*)d_in[5];
    const float* b2   = (const float*)d_in[6];

    char* ws = (char*)d_ws;
    unsigned short* w1bf   = (unsigned short*)ws;            // 2 MB
    float*          scores = (float*)(ws + (2u << 20));      // 1 MB
    int*            idxWs  = (int*)  (ws + (3u << 20));      // 64 KB

    float* out       = (float*)d_out;
    float* outEmb    = out;                                   // B*K*D
    float* outMask   = outEmb    + (size_t)B_ * K_ * D_;
    float* outIdx    = outMask   + (size_t)B_ * K_;
    float* outScores = outIdx    + (size_t)B_ * K_;
    float* outNk     = outScores + (size_t)B_ * K_;

    hipLaunchKernelGGL(convert_w1_kernel, dim3((D_ * H_) / 256), dim3(256), 0, stream,
                       W1, w1bf);
    hipLaunchKernelGGL(scorer_kernel, dim3((B_ * N_) / 32), dim3(256), 0, stream,
                       emb, mask, b1, w2, b2, w1bf, scores);
    hipLaunchKernelGGL(topk_kernel, dim3(B_), dim3(512), 0, stream,
                       scores, mask, outMask, outIdx, outScores, outNk, idxWs);
    hipLaunchKernelGGL(gather_kernel, dim3(B_ * K_), dim3(256), 0, stream,
                       emb, idxWs, outEmb);
}
